// FASTMultiHeadAttention_6158983102981
// MI455X (gfx1250) — matrix operand order repaired and checked by reading, not compiled, not run on hardware
//
#include <hip/hip_runtime.h>

// ---------------------------------------------------------------------------
// fastmax attention (polynomial softmax, causal, RPE) for MI455X / gfx1250
// B=2 H=16 N=1024 D=64, fp32 in/out, bf16 WMMA compute (f32 accumulate)
// ---------------------------------------------------------------------------

typedef __attribute__((ext_vector_type(16))) __bf16 v16bf;
typedef __attribute__((ext_vector_type(8)))  __bf16 v8bf;
typedef __attribute__((ext_vector_type(8)))  float  v8f;
typedef __attribute__((ext_vector_type(4)))  float  fv4;

#define B_  2
#define H_  16
#define N_  1024
#define D_  64

#define KS 72   // kbuf row stride (bf16)
#define VS 40   // vtbuf row stride (bf16)

static __device__ __forceinline__ v8f wmma_bf16(v16bf a, v16bf b, v8f c) {
  // D = A(16x32 bf16) x B(32x16 bf16) + C(16x16 f32)
  return __builtin_amdgcn_wmma_f32_16x16x32_bf16(false, a, false, b,
                                                 (short)0, c, false, false);
}

// wave-local LDS ordering (stores by this wave complete before later reads)
static __device__ __forceinline__ void lds_wave_fence() {
  asm volatile("s_wait_dscnt 0" ::: "memory");
}

__global__ __launch_bounds__(128)
void fastmax_attn(const float* __restrict__ Q, const float* __restrict__ K,
                  const float* __restrict__ V, const float* __restrict__ RPE,
                  float* __restrict__ O)
{
  __shared__ __bf16 kbuf[32 * KS];      // K tile, row-major [j][d] bf16
  __shared__ __bf16 vtbuf[64 * VS];     // V tile transposed [d][j] bf16
  __shared__ float  tshare[4][2][256];  // per-wave T tiles, [u][m] f32
  __shared__ float  pbuf[4][32 * 16];   // per-wave P tile, [j][m] f32

  const int tid  = threadIdx.x;
  const int lane = tid & 31;
  const int wv   = tid >> 5;     // wave id 0..3
  const int lm   = lane & 15;
  const int lh   = lane >> 4;

  const int bh  = blockIdx.y;
  const int i0b = blockIdx.x * 64;
  const int i0w = i0b + wv * 16;       // this wave's query strip base

  const size_t base = (size_t)bh * N_ * D_;
  const float* Qb = Q + base;
  const float* Kb = K + base;
  const float* Vb = V + base;
  float*       Ob = O + base;
  const float* R2 = RPE + (size_t)(N_ - 1) * D_;   // R2[t][d], t >= 0

  // ---- Q strip -> A-matrix registers (two 32-wide K chunks) ----
  // A operand, contiguous k order used here (B matches): lane m=lm; elems 0..15 -> K=16*lh+e
  v16bf qa[2];
  {
    const float* qr = Qb + (size_t)(i0w + lm) * D_;
#pragma unroll
    for (int c = 0; c < 2; ++c) {
      fv4 a0 = *(const fv4*)(qr + 32 * c + 16 * lh);
      fv4 a1 = *(const fv4*)(qr + 32 * c + 16 * lh + 4);
      fv4 a2 = *(const fv4*)(qr + 32 * c + 16 * lh + 8);
      fv4 a3 = *(const fv4*)(qr + 32 * c + 16 * lh + 12);
      v16bf a;
#pragma unroll
      for (int e = 0; e < 4; ++e) {
        a[e]      = (__bf16)a0[e];
        a[4 + e]  = (__bf16)a1[e];
        a[8 + e]  = (__bf16)a2[e];
        a[12 + e] = (__bf16)a3[e];
      }
      qa[c] = a;
    }
  }

  // T tile: Q_strip(16xD) x R2[tb..tb+16)^T  -> 16x16 f32 (C layout)
  auto computeT = [&](int tb) -> v8f {
    v8f acc = {0, 0, 0, 0, 0, 0, 0, 0};
    const float* rr = R2 + (size_t)(tb + lm) * D_;   // B col u = lm
#pragma unroll
    for (int c = 0; c < 2; ++c) {
      fv4 b0 = *(const fv4*)(rr + 32 * c + 16 * lh);
      fv4 b1 = *(const fv4*)(rr + 32 * c + 16 * lh + 4);
      fv4 b2 = *(const fv4*)(rr + 32 * c + 16 * lh + 8);
      fv4 b3 = *(const fv4*)(rr + 32 * c + 16 * lh + 12);
      v16bf b;
#pragma unroll
      for (int e = 0; e < 4; ++e) {
        b[e]      = (__bf16)b0[e];
        b[4 + e]  = (__bf16)b1[e];
        b[8 + e]  = (__bf16)b2[e];
        b[12 + e] = (__bf16)b3[e];
      }
      acc = wmma_bf16(qa[c], b, acc);
    }
    return acc;
  };

  // store C-layout tile as [u][m] f32 (consecutive in vgpr index -> 2x b128)
  auto storeT = [&](float* buf, v8f t) {
    float* p = buf + lm * 16 + 8 * lh;
    fv4 lo = {t[0], t[1], t[2], t[3]};
    fv4 hi = {t[4], t[5], t[6], t[7]};
    *(fv4*)p = lo;
    *(fv4*)(p + 4) = hi;
  };

  // ---- init sliding T window: tshare[cur]=base i0w, tshare[cur^1]=base i0w-16
  int cur = 0;
  {
    storeT(tshare[wv][0], computeT(i0w));
    v8f t1 = {0, 0, 0, 0, 0, 0, 0, 0};
    if (i0w >= 16) t1 = computeT(i0w - 16);
    storeT(tshare[wv][1], t1);
  }

  v8f oacc[4];
#pragma unroll
  for (int r = 0; r < 4; ++r) oacc[r] = (v8f){0, 0, 0, 0, 0, 0, 0, 0};
  v8f dacc = {0, 0, 0, 0, 0, 0, 0, 0};
  v16bf ones;
#pragma unroll
  for (int e = 0; e < 16; ++e) ones[e] = (__bf16)1.0f;

  const int nsteps = i0b / 32 + 2;     // causal: keys [0, i0b+64)
  for (int st = 0; st < nsteps; ++st) {
    const int j0 = st * 32;
    __syncthreads();   // previous tile fully consumed

    // ---- cooperative K tile load: 32 rows x 64 d, f32 -> bf16 LDS ----
    {
      const int row = tid >> 2;
      const int qd  = (tid & 3) * 16;
      const float* kr = Kb + (size_t)(j0 + row) * D_ + qd;
      fv4 x0 = *(const fv4*)(kr);
      fv4 x1 = *(const fv4*)(kr + 4);
      fv4 x2 = *(const fv4*)(kr + 8);
      fv4 x3 = *(const fv4*)(kr + 12);
      v8bf w0, w1;
#pragma unroll
      for (int e = 0; e < 4; ++e) {
        w0[e]     = (__bf16)x0[e];
        w0[4 + e] = (__bf16)x1[e];
        w1[e]     = (__bf16)x2[e];
        w1[4 + e] = (__bf16)x3[e];
      }
      *(v8bf*)(kbuf + row * KS + qd)     = w0;
      *(v8bf*)(kbuf + row * KS + qd + 8) = w1;
    }
    // ---- cooperative V tile load + transpose: vtbuf[d][j] bf16 ----
    {
      const int p = tid & 15;     // row pair (2p, 2p+1)
      const int g = tid >> 4;     // d group of 8
      const float* v0 = Vb + (size_t)(j0 + 2 * p) * D_ + 8 * g;
      const float* v1 = v0 + D_;
      fv4 a0 = *(const fv4*)(v0);
      fv4 a1 = *(const fv4*)(v0 + 4);
      fv4 b0 = *(const fv4*)(v1);
      fv4 b1 = *(const fv4*)(v1 + 4);
#pragma unroll
      for (int e = 0; e < 8; ++e) {
        float fa = (e < 4) ? a0[e] : a1[e - 4];
        float fb = (e < 4) ? b0[e] : b1[e - 4];
        __bf16* dst = vtbuf + (8 * g + e) * VS + 2 * p;
        dst[0] = (__bf16)fa;
        dst[1] = (__bf16)fb;
      }
    }
    // prefetch next K/V tiles into caches
    if (st + 1 < nsteps) {
      const int row = tid >> 2;
      const int qd  = (tid & 3) * 16;
      __builtin_prefetch(Kb + (size_t)(j0 + 32 + row) * D_ + qd, 0, 1);
      __builtin_prefetch(Vb + (size_t)(j0 + 32 + row) * D_ + qd, 0, 1);
    }
    __syncthreads();   // tiles visible to all waves

    if (j0 <= i0w + 15) {
#pragma unroll
      for (int sub = 0; sub < 2; ++sub) {
        const int j0s = j0 + sub * 16;
        if (j0s > i0w + 15) {
          // fully masked subtile: zero its P columns
          float* pp = pbuf[wv] + (sub * 16) * 16;
#pragma unroll
          for (int e = 0; e < 8; ++e) pp[(8 * lh + e) * 16 + lm] = 0.0f;
          continue;
        }
        // ---- S = Q x K^T (16x16, 2 WMMAs over D) ----
        v8f s = {0, 0, 0, 0, 0, 0, 0, 0};
        const __bf16* krow = kbuf + (sub * 16 + lm) * KS;  // B col n = lm
#pragma unroll
        for (int c = 0; c < 2; ++c) {
          v16bf b = *(const v16bf*)(krow + 32 * c + 16 * lh);
          s = wmma_bf16(qa[c], b, s);
        }
        // ---- RPE bias gather + polynomial score ----
        const int d0 = i0w - j0s;                  // 16-aligned, >= 0
        const float* tc = tshare[wv][cur];         // covers t in [d0, d0+16)
        const float* tp = tshare[wv][cur ^ 1];     // covers t in [d0-16, d0)
        float* ps = pbuf[wv] + (sub * 16 + lm) * 16 + 8 * lh;
        fv4 outlo, outhi;
#pragma unroll
        for (int vv = 0; vv < 8; ++vv) {
          const int m = vv + 8 * lh;
          const int t = d0 + m - lm;               // relative offset i-j
          float bias;
          if (t >= d0) bias = tc[(t - d0) * 16 + m];
          else         bias = tp[(t - d0 + 16) * 16 + m];
          const float sv = s[vv] + bias;
          float sc = 1.0f + sv + 0.5f * sv * sv;   // A0 + A1*s + A2*s^2
          if (t < 0) sc = 0.0f;                    // causal mask
          if (vv < 4) outlo[vv] = sc; else outhi[vv - 4] = sc;
        }
        *(fv4*)ps       = outlo;
        *(fv4*)(ps + 4) = outhi;
        // ---- slide T window: compute tile at base d0-32 ----
        v8f tn = {0, 0, 0, 0, 0, 0, 0, 0};
        if (d0 - 32 >= 0) tn = computeT(d0 - 32);
        storeT((float*)tshare[wv][cur], tn);       // overwrite oldest
        cur ^= 1;
        lds_wave_fence();
      }

      lds_wave_fence();   // P + T stores visible to all lanes of this wave
      // ---- P (16x32) back as A-matrix ----
      v16bf pa;
#pragma unroll
      for (int e = 0; e < 8; ++e) {
        pa[e]     = (__bf16)pbuf[wv][(16 * lh + e) * 16 + lm];
        pa[8 + e] = (__bf16)pbuf[wv][(16 * lh + 8 + e) * 16 + lm];
      }
      // denominator: row sums replicated into C layout
      dacc = wmma_bf16(pa, ones, dacc);
      // O += P x V (K-dim = 32 keys, 4 d-tiles of 16)
#pragma unroll
      for (int r = 0; r < 4; ++r) {
        const __bf16* vr = vtbuf + (16 * r + lm) * VS + 16 * lh;
        v16bf b = *(const v16bf*)vr;
        oacc[r] = wmma_bf16(pa, b, oacc[r]);
      }
    }
  }

  // ---- epilogue: O = O_num / denom (denom replicated per lane slot) ----
#pragma unroll
  for (int vv = 0; vv < 8; ++vv) {
    const float inv = 1.0f / dacc[vv];
    float* orow = Ob + (size_t)(i0w + vv + 8 * lh) * D_ + lm;
#pragma unroll
    for (int r = 0; r < 4; ++r) orow[16 * r] = oacc[r][vv] * inv;
  }
}

extern "C" void kernel_launch(void* const* d_in, const int* in_sizes, int n_in,
                              void* d_out, int out_size, void* d_ws, size_t ws_size,
                              hipStream_t stream) {
  const float* q   = (const float*)d_in[0];
  const float* k   = (const float*)d_in[1];
  const float* v   = (const float*)d_in[2];
  const float* rpe = (const float*)d_in[4];   // d_in[3] = drop_noise (unused)
  float* out = (float*)d_out;
  (void)in_sizes; (void)n_in; (void)out_size; (void)d_ws; (void)ws_size;

  dim3 grid(N_ / 64, B_ * H_);   // (row blocks, b*h)
  fastmax_attn<<<grid, 128, 0, stream>>>(q, k, v, rpe, out);
}
